// ImprovedEdgeGraphSAGE_44444321579081
// MI455X (gfx1250) — matrix operand order repaired and checked by reading, not compiled, not run on hardware
//
#include <hip/hip_runtime.h>
#include <cstddef>

// ---------------------------------------------------------------------------
// MI455X / gfx1250, wave32. All matrix math via v_wmma_f32_16x16x32_bf16.
// Fragments are loaded from LDS with contiguous 16B (b128) reads:
//   A-frag lane layout: rows = lane%16, k = {half*8+0..7, 16+half*8+0..7}
//   B-frag lane layout: col  = lane%16, k = {half*8+0..7, 16+half*8+0..7} (same as A)
// so B tiles are staged n-major and V tiles transposed (d-major).
// ---------------------------------------------------------------------------

typedef __attribute__((ext_vector_type(16))) __bf16 v16bf;
typedef __attribute__((ext_vector_type(8)))  __bf16 v8bf;
typedef __attribute__((ext_vector_type(4)))  __bf16 v4bf;
typedef __attribute__((ext_vector_type(8)))  float  v8f;

#define NNODES 1024
#define DEG    4
#define NEDGES 4096
#define HID    256
#define ATTND  512
#define DH     128
#define NHEADS 4

__device__ __forceinline__ __bf16 tobf(float f) { return (__bf16)f; }
__device__ __forceinline__ float  sigf(float x) { return 1.0f / (1.0f + __expf(-x)); }

__device__ __forceinline__ v16bf cat16(v8bf a, v8bf b) {
    return __builtin_shufflevector(a, b, 0, 1, 2, 3, 4, 5, 6, 7,
                                   8, 9, 10, 11, 12, 13, 14, 15);
}
__device__ __forceinline__ v4bf pk4(float4 v) {
    v4bf o; o[0] = tobf(v.x); o[1] = tobf(v.y); o[2] = tobf(v.z); o[3] = tobf(v.w);
    return o;
}

// ---------------------------------------------------------------------------
// Generic GEMM: C[M,N] = A[M,K] @ W[N,K]^T  (+C if flags&1) (+bias) (+resid)
// (relu if flags&2).  Optional row gather on A: a_row = gidx[m*gmul+goff].
// REQUIRES: M % 64 == 0, K % 32 == 0 (true for every call site). N tail OK.
// Block: 128 threads = 4 waves, tile 64(M) x 16(N), K step 32.
// All row pointers (incl. the gather) are hoisted out of the K loop.
// ---------------------------------------------------------------------------
__global__ __launch_bounds__(128) void gemm_bf16_wmma(
    const float* __restrict__ A, int lda,
    const int* __restrict__ gidx, int gmul, int goff,
    const float* __restrict__ W,
    const float* __restrict__ bias,
    const float* __restrict__ resid,
    float* __restrict__ C, int ldc,
    int M, int N, int K, int flags)
{
    __shared__ __bf16 As[64][32];   // row-major A tile, 64B row pitch
    __shared__ __bf16 Bs[16][32];   // n-major W tile (B[k][n] = Bs[n][k])

    const int wave  = threadIdx.x >> 5;
    const int lane  = threadIdx.x & 31;
    const int lhalf = lane >> 4;
    const int l16   = lane & 15;
    const int m0 = blockIdx.x * 64;
    const int n0 = blockIdx.y * 16;

    // ---- hoisted per-thread staging addresses (loop-invariant) ----
    const float* arow[4];
    __bf16* asd[4];
    #pragma unroll
    for (int i = 0; i < 4; ++i) {
        int e = i * 512 + threadIdx.x * 4;
        int r = e >> 5, c = e & 31;
        int row = gidx ? gidx[(m0 + r) * gmul + goff] : (m0 + r);
        arow[i] = A + (size_t)row * lda + c;
        asd[i]  = &As[r][c];
    }
    const int  wn   = (threadIdx.x * 4) >> 5;
    const int  wc   = (threadIdx.x * 4) & 31;
    const bool wok  = (n0 + wn) < N;
    const float* wrow = W + (size_t)(n0 + (wok ? wn : 0)) * K + wc;
    __bf16* bsd = &Bs[wn][wc];
    v4bf zero4; zero4[0] = zero4[1] = zero4[2] = zero4[3] = tobf(0.f);

    // ---- hoisted fragment LDS read addresses ----
    const v8bf* afp0 = (const v8bf*)&As[wave * 16 + l16][lhalf * 8];
    const v8bf* afp1 = (const v8bf*)&As[wave * 16 + l16][16 + lhalf * 8];
    const v8bf* bfp0 = (const v8bf*)&Bs[l16][lhalf * 8];
    const v8bf* bfp1 = (const v8bf*)&Bs[l16][16 + lhalf * 8];

    v8f acc;
    #pragma unroll
    for (int r = 0; r < 8; ++r) acc[r] = 0.0f;

    for (int k0 = 0; k0 < K; k0 += 32) {
        // stage A tile 64x32: 4 x float4 per thread, unguarded
        #pragma unroll
        for (int i = 0; i < 4; ++i) {
            float4 v = *(const float4*)(arow[i] + k0);
            __builtin_prefetch(arow[i] + k0 + 32, 0, 0); // global_prefetch_b8
            *(v4bf*)asd[i] = pk4(v);
        }
        // stage W tile 16x32 n-major: 1 x float4 per thread, N-guarded
        if (wok) {
            float4 v = *(const float4*)(wrow + k0);
            __builtin_prefetch(wrow + k0 + 32, 0, 0);
            *(v4bf*)bsd = pk4(v);
        } else {
            *(v4bf*)bsd = zero4;
        }
        __syncthreads();

        // fragments: contiguous b128 LDS reads
        v16bf af = cat16(*afp0, *afp1);
        v16bf bf = cat16(*bfp0, *bfp1);
        acc = __builtin_amdgcn_wmma_f32_16x16x32_bf16(false, af, false, bf,
                                                      (short)0, acc, false, false);
        __syncthreads();
    }

    // epilogue (C layout: row = vgpr + 8*half, col = lane%16)
    #pragma unroll
    for (int r = 0; r < 8; ++r) {
        int m = m0 + wave * 16 + lhalf * 8 + r;
        int n = n0 + l16;
        if (n < N) {
            size_t idx = (size_t)m * ldc + n;
            float v = acc[r];
            if (flags & 1) v += C[idx];
            if (bias)      v += bias[n];
            if (resid)     v += resid[idx];
            if (flags & 2) v = fmaxf(v, 0.0f);
            C[idx] = v;
        }
    }
}

// ---------------------------------------------------------------------------
// Flash attention over E=4096 edge tokens, 4 heads, dh=128.
// Block = 128 threads (4 waves), 64 q-rows per block, 32 keys per iteration.
// qkv layout per row: [q(512) | k(512) | v(512)], head h uses cols h*128..+127.
// ---------------------------------------------------------------------------
__global__ __launch_bounds__(128) void flash_attn_kernel(
    const float* __restrict__ qkv, float* __restrict__ out)
{
    __shared__ __bf16 Ks[32][DH + 8];  // key-major (row pitch 272B, 16B aligned)
    __shared__ __bf16 Vt[DH][32];      // d-major (transposed V tile)
    __shared__ __bf16 Ps[4][16][32];   // per-wave P scratch, 64B row pitch

    const int head  = blockIdx.y;
    const int q0    = blockIdx.x * 64;
    const int wave  = threadIdx.x >> 5;
    const int lane  = threadIdx.x & 31;
    const int lhalf = lane >> 4;
    const int l16   = lane & 15;

    const float* Qp = qkv + head * DH;
    const float scale = 0.08838834764831845f; // 1/sqrt(128), folded into Q

    const int qrow = q0 + wave * 16 + l16;

    // Q as 4 bf16 A-fragments (16 rows x 128 d), scale folded in.
    v16bf qf[4];
    #pragma unroll
    for (int c = 0; c < 4; ++c) {
        #pragma unroll
        for (int g = 0; g < 2; ++g) {
            const float* p = Qp + (size_t)qrow * (3 * ATTND) + c * 32 + g * 16 + lhalf * 8;
            float4 x0 = *(const float4*)p;
            float4 x1 = *(const float4*)(p + 4);
            qf[c][g * 8 + 0] = tobf(x0.x * scale);
            qf[c][g * 8 + 1] = tobf(x0.y * scale);
            qf[c][g * 8 + 2] = tobf(x0.z * scale);
            qf[c][g * 8 + 3] = tobf(x0.w * scale);
            qf[c][g * 8 + 4] = tobf(x1.x * scale);
            qf[c][g * 8 + 5] = tobf(x1.y * scale);
            qf[c][g * 8 + 6] = tobf(x1.z * scale);
            qf[c][g * 8 + 7] = tobf(x1.w * scale);
        }
    }

    // hoisted staging pointers, advanced by 32 keys per iteration
    const float* kstg[8];
    __bf16*      kdst[8];
    const float* vstg[8];
    __bf16*      vdst[8];
    #pragma unroll
    for (int i = 0; i < 8; ++i) {
        int e  = i * 512 + threadIdx.x * 4;
        int kk = e >> 7, d = e & 127;
        kstg[i] = qkv + ATTND + head * DH + (size_t)kk * (3 * ATTND) + d;
        kdst[i] = &Ks[kk][d];
        int dv = e >> 5, kv = e & 31;
        vstg[i] = qkv + 2 * ATTND + head * DH + (size_t)kv * (3 * ATTND) + dv;
        vdst[i] = &Vt[dv][kv];
    }

    // hoisted fragment LDS read addresses
    const v8bf* pap0 = (const v8bf*)&Ps[wave][l16][lhalf * 8];
    const v8bf* pap1 = (const v8bf*)&Ps[wave][l16][16 + lhalf * 8];

    float mrow[8], lrow[8];
    v8f of[8];
    #pragma unroll
    for (int r = 0; r < 8; ++r) { mrow[r] = -1e30f; lrow[r] = 0.0f; }
    #pragma unroll
    for (int dt = 0; dt < 8; ++dt)
        #pragma unroll
        for (int r = 0; r < 8; ++r) of[dt][r] = 0.0f;

    for (int kt = 0; kt < NEDGES; kt += 32) {
        __syncthreads(); // prior iteration done with Ks/Vt/Ps
        // stage K (key-major): 8 x float4 per thread
        #pragma unroll
        for (int i = 0; i < 8; ++i) {
            float4 v = *(const float4*)kstg[i];
            __builtin_prefetch(kstg[i] + 32 * 3 * ATTND, 0, 0);
            *(v4bf*)kdst[i] = pk4(v);
            kstg[i] += 32 * 3 * ATTND;
        }
        // stage V transposed (d-major): 4 keys per v4bf store
        #pragma unroll
        for (int i = 0; i < 8; ++i) {
            v4bf o;
            #pragma unroll
            for (int u = 0; u < 4; ++u)
                o[u] = tobf(vstg[i][(size_t)u * (3 * ATTND)]);
            *(v4bf*)vdst[i] = o;
            vstg[i] += 32 * 3 * ATTND;
        }
        __syncthreads();

        // S = Q @ K^T : two 16x16 n-tiles (keys 0-15, 16-31), contract over d
        v8f s0, s1;
        #pragma unroll
        for (int r = 0; r < 8; ++r) { s0[r] = 0.0f; s1[r] = 0.0f; }
        #pragma unroll
        for (int c = 0; c < 4; ++c) {
            int dco = c * 32 + lhalf * 8;
            v16bf b0 = cat16(*(const v8bf*)&Ks[l16][dco],
                             *(const v8bf*)&Ks[l16][dco + 16]);
            v16bf b1 = cat16(*(const v8bf*)&Ks[l16 + 16][dco],
                             *(const v8bf*)&Ks[l16 + 16][dco + 16]);
            s0 = __builtin_amdgcn_wmma_f32_16x16x32_bf16(false, qf[c], false, b0,
                                                         (short)0, s0, false, false);
            s1 = __builtin_amdgcn_wmma_f32_16x16x32_bf16(false, qf[c], false, b1,
                                                         (short)0, s1, false, false);
        }

        // online softmax: rows live in lanes of one half (shfl_xor 1/2/4/8)
        float scl[8];
        #pragma unroll
        for (int r = 0; r < 8; ++r) {
            float mx = fmaxf(s0[r], s1[r]);
            mx = fmaxf(mx, __shfl_xor(mx, 1, 32));
            mx = fmaxf(mx, __shfl_xor(mx, 2, 32));
            mx = fmaxf(mx, __shfl_xor(mx, 4, 32));
            mx = fmaxf(mx, __shfl_xor(mx, 8, 32));
            float mn = fmaxf(mrow[r], mx);
            float e0 = __expf(s0[r] - mn);
            float e1 = __expf(s1[r] - mn);
            float sum = e0 + e1;
            sum += __shfl_xor(sum, 1, 32);
            sum += __shfl_xor(sum, 2, 32);
            sum += __shfl_xor(sum, 4, 32);
            sum += __shfl_xor(sum, 8, 32);
            scl[r]  = __expf(mrow[r] - mn);
            lrow[r] = lrow[r] * scl[r] + sum;
            mrow[r] = mn;
            s0[r] = e0; s1[r] = e1;
        }
        #pragma unroll
        for (int dt = 0; dt < 8; ++dt)
            #pragma unroll
            for (int r = 0; r < 8; ++r) of[dt][r] *= scl[r];

        // restage P (16x32) through LDS into A-fragment layout
        #pragma unroll
        for (int r = 0; r < 8; ++r) {
            int row = r + 8 * lhalf;
            Ps[wave][row][l16]      = tobf(s0[r]);
            Ps[wave][row][l16 + 16] = tobf(s1[r]);
        }
        __syncthreads();

        v16bf pa = cat16(*pap0, *pap1);
        #pragma unroll
        for (int dt = 0; dt < 8; ++dt) {
            v16bf vb = cat16(*(const v8bf*)&Vt[dt * 16 + l16][lhalf * 8],
                             *(const v8bf*)&Vt[dt * 16 + l16][16 + lhalf * 8]);
            of[dt] = __builtin_amdgcn_wmma_f32_16x16x32_bf16(false, pa, false, vb,
                                                             (short)0, of[dt], false, false);
        }
    }

    // epilogue: o[q, h*128 + d] = O / l
    #pragma unroll
    for (int dt = 0; dt < 8; ++dt)
        #pragma unroll
        for (int r = 0; r < 8; ++r) {
            int row = q0 + wave * 16 + lhalf * 8 + r;
            int d   = dt * 16 + l16;
            out[(size_t)row * ATTND + head * DH + d] = of[dt][r] / lrow[r];
        }
}

// ---------------------------------------------------------------------------
// Elementwise kernels
// ---------------------------------------------------------------------------
__global__ void fill0_kernel(float* __restrict__ p, int n) {
    int i = blockIdx.x * blockDim.x + threadIdx.x;
    if (i < n) p[i] = 0.0f;
}

// PyTorch LSTM cell: gates [N,4F] in i,f,g,o order; updates h,c in place
__global__ void lstm_cell_kernel(const float* __restrict__ gates,
                                 float* __restrict__ h, float* __restrict__ c,
                                 int n, int F) {
    int i = blockIdx.x * blockDim.x + threadIdx.x;
    if (i >= n * F) return;
    int row = i / F, col = i - row * F;
    const float* g = gates + (size_t)row * 4 * F;
    float ig = sigf(g[col]);
    float fg = sigf(g[F + col]);
    float gg = tanhf(g[2 * F + col]);
    float og = sigf(g[3 * F + col]);
    float cn = fg * c[i] + ig * gg;
    c[i] = cn;
    h[i] = og * tanhf(cn);
}

// ef[e] = concat(h[src[e]], h[dst[e]])   -> [E, 512]
__global__ void build_ef_kernel(const float* __restrict__ h,
                                const int* __restrict__ src,
                                const int* __restrict__ dst,
                                float* __restrict__ ef) {
    int i = blockIdx.x * blockDim.x + threadIdx.x;
    if (i >= NEDGES * 512) return;
    int e = i >> 9, c = i & 511;
    ef[i] = (c < HID) ? h[(size_t)src[e] * HID + c]
                      : h[(size_t)dst[e] * HID + (c - HID)];
}

// z0[e] = concat(att_out[e, 256:512], edge_attr[e])  -> [E, 288]
__global__ void build_z0_kernel(const float* __restrict__ att_out,
                                const float* __restrict__ edge_attr,
                                float* __restrict__ z0) {
    int i = blockIdx.x * blockDim.x + threadIdx.x;
    if (i >= NEDGES * 288) return;
    int e = i / 288, c = i - e * 288;
    z0[i] = (c < HID) ? att_out[(size_t)e * ATTND + (ATTND - HID) + c]
                      : edge_attr[(size_t)e * 32 + (c - HID)];
}

// ---------------------------------------------------------------------------
// Host orchestration
// ---------------------------------------------------------------------------
extern "C" void kernel_launch(void* const* d_in, const int* in_sizes, int n_in,
                              void* d_out, int out_size, void* d_ws, size_t ws_size,
                              hipStream_t stream) {
    (void)in_sizes; (void)n_in; (void)out_size; (void)ws_size;

    const float* x         = (const float*)d_in[0];
    const int*   ei        = (const int*)  d_in[1];
    const float* edge_attr = (const float*)d_in[2];
    const float* l1_wih = (const float*)d_in[3];
    const float* l1_whh = (const float*)d_in[4];
    const float* l1_bih = (const float*)d_in[5];
    const float* l1_bhh = (const float*)d_in[6];
    const float* l1_wl  = (const float*)d_in[7];
    const float* l1_bl  = (const float*)d_in[8];
    const float* l1_wr  = (const float*)d_in[9];
    const float* ls_wih = (const float*)d_in[10];
    const float* ls_whh = (const float*)d_in[11];
    const float* ls_bih = (const float*)d_in[12];
    const float* ls_bhh = (const float*)d_in[13];
    const float* ls_wl  = (const float*)d_in[14];
    const float* ls_bl  = (const float*)d_in[15];
    const float* ls_wr  = (const float*)d_in[16];
    const float* attn_in_w  = (const float*)d_in[17];
    const float* attn_in_b  = (const float*)d_in[18];
    const float* attn_out_w = (const float*)d_in[19];
    const float* attn_out_b = (const float*)d_in[20];
    const float* mlp_w1 = (const float*)d_in[21];
    const float* mlp_b1 = (const float*)d_in[22];
    const float* mlp_w2 = (const float*)d_in[23];
    const float* mlp_b2 = (const float*)d_in[24];
    const float* mlp_w3 = (const float*)d_in[25];
    const float* mlp_b3 = (const float*)d_in[26];

    const int* src = ei;
    const int* dst = ei + NEDGES;

    // scratch layout (floats), qkv region reused for att_out/z* after attention
    float* ws = (float*)d_ws;
    size_t off = 0;
    auto alloc = [&](size_t n) { float* p = ws + off; off += n; return p; };
    float* h0     = alloc((size_t)NNODES * HID);
    float* h1     = alloc((size_t)NNODES * HID);
    float* lh     = alloc((size_t)NNODES * HID);
    float* lc     = alloc((size_t)NNODES * HID);
    float* gates  = alloc((size_t)NNODES * 4 * HID);
    float* ef     = alloc((size_t)NEDGES * ATTND);
    float* attn_o = alloc((size_t)NEDGES * ATTND);
    float* qkv    = alloc((size_t)NEDGES * 3 * ATTND);
    float* att_out = qkv;                          // reuse after flash attention
    float* z0 = att_out + (size_t)NEDGES * ATTND;  // [E,288]
    float* z1 = z0 + (size_t)NEDGES * 288;         // [E,256]
    float* z2 = z1 + (size_t)NEDGES * HID;         // [E,128]

    auto gemm = [&](const float* A, int lda, const int* gidx, int gmul, int goff,
                    const float* W, const float* bias, const float* resid,
                    float* C, int ldc, int M, int Nn, int K, int flags) {
        dim3 grid((M + 63) / 64, (Nn + 15) / 16);
        gemm_bf16_wmma<<<grid, 128, 0, stream>>>(A, lda, gidx, gmul, goff, W, bias,
                                                 resid, C, ldc, M, Nn, K, flags);
    };
    auto fill = [&](float* p, int n) {
        fill0_kernel<<<(n + 255) / 256, 256, 0, stream>>>(p, n);
    };

    // ---- conv1: SAGE(128 -> 256), LSTM aggregation over DEG=4 neighbors ----
    fill(lh, NNODES * 128);
    fill(lc, NNODES * 128);
    for (int t = 0; t < DEG; ++t) {
        // gates = x[src[:,t]] @ wih^T + bih
        gemm(x, 128, src, DEG, t, l1_wih, l1_bih, nullptr, gates, 512,
             NNODES, 512, 128, 0);
        // gates += lh @ whh^T + bhh
        gemm(lh, 128, nullptr, 0, 0, l1_whh, l1_bhh, nullptr, gates, 512,
             NNODES, 512, 128, 1);
        lstm_cell_kernel<<<(NNODES * 128 + 255) / 256, 256, 0, stream>>>(
            gates, lh, lc, NNODES, 128);
    }
    // h0 = relu(agg @ wl^T + bl + x @ wr^T)
    gemm(lh, 128, nullptr, 0, 0, l1_wl, l1_bl, nullptr, h0, HID, NNODES, HID, 128, 0);
    gemm(x, 128, nullptr, 0, 0, l1_wr, nullptr, nullptr, h0, HID, NNODES, HID, 128, 1 | 2);

    // ---- conv2..conv4: SAGE(256 -> 256) + residual ----
    float* hin = h0;
    float* hout = h1;
    for (int i = 0; i < 3; ++i) {
        const float* wih = ls_wih + (size_t)i * 4 * HID * HID;
        const float* whh = ls_whh + (size_t)i * 4 * HID * HID;
        const float* bih = ls_bih + (size_t)i * 4 * HID;
        const float* bhh = ls_bhh + (size_t)i * 4 * HID;
        const float* wl  = ls_wl  + (size_t)i * HID * HID;
        const float* bl  = ls_bl  + (size_t)i * HID;
        const float* wr  = ls_wr  + (size_t)i * HID * HID;

        fill(lh, NNODES * HID);
        fill(lc, NNODES * HID);
        for (int t = 0; t < DEG; ++t) {
            gemm(hin, HID, src, DEG, t, wih, bih, nullptr, gates, 4 * HID,
                 NNODES, 4 * HID, HID, 0);
            gemm(lh, HID, nullptr, 0, 0, whh, bhh, nullptr, gates, 4 * HID,
                 NNODES, 4 * HID, HID, 1);
            lstm_cell_kernel<<<(NNODES * HID + 255) / 256, 256, 0, stream>>>(
                gates, lh, lc, NNODES, HID);
        }
        gemm(lh, HID, nullptr, 0, 0, wl, bl, nullptr, hout, HID, NNODES, HID, HID, 0);
        gemm(hin, HID, nullptr, 0, 0, wr, nullptr, hin, hout, HID, NNODES, HID, HID, 1 | 2);
        float* tmp = hin; hin = hout; hout = tmp;
    }
    // hin now holds final node features [N, 256]

    // ---- edge features + attention ----
    build_ef_kernel<<<(NEDGES * 512 + 255) / 256, 256, 0, stream>>>(hin, src, dst, ef);
    gemm(ef, ATTND, nullptr, 0, 0, attn_in_w, attn_in_b, nullptr, qkv, 3 * ATTND,
         NEDGES, 3 * ATTND, ATTND, 0);
    flash_attn_kernel<<<dim3(NEDGES / 64, NHEADS), 128, 0, stream>>>(qkv, attn_o);
    gemm(attn_o, ATTND, nullptr, 0, 0, attn_out_w, attn_out_b, nullptr, att_out, ATTND,
         NEDGES, ATTND, ATTND, 0);

    // ---- edge MLP ----
    build_z0_kernel<<<(NEDGES * 288 + 255) / 256, 256, 0, stream>>>(att_out, edge_attr, z0);
    gemm(z0, 288, nullptr, 0, 0, mlp_w1, mlp_b1, nullptr, z1, HID, NEDGES, HID, 288, 2);
    gemm(z1, HID, nullptr, 0, 0, mlp_w2, mlp_b2, nullptr, z2, 128, NEDGES, 128, HID, 2);
    gemm(z2, 128, nullptr, 0, 0, mlp_w3, mlp_b3, nullptr, (float*)d_out, 2,
         NEDGES, 2, 128, 0);
}
